// SelfAttention_18056042512616
// MI455X (gfx1250) — matrix operand order repaired and checked by reading, not compiled, not run on hardware
//
#include <hip/hip_runtime.h>
#include <hip/hip_bf16.h>

// ---------------- problem constants ----------------
#define HEADS  8
#define DMODEL 512
#define DK     64
#define BATCH  4
#define SEQ    2048
#define BN     (BATCH * SEQ)   // 8192 total token rows

typedef __attribute__((ext_vector_type(16))) _Float16 v16h;
typedef __attribute__((ext_vector_type(8)))  _Float16 v8h;
typedef __attribute__((ext_vector_type(8)))  float    v8f;

static __device__ __forceinline__ v8f vzero8() {
  v8f z = {0.f, 0.f, 0.f, 0.f, 0.f, 0.f, 0.f, 0.f};
  return z;
}

// D = A(16x32 f16) * B(32x16 f16) + C, f32 accumulate
static __device__ __forceinline__ v8f wmma_f16(v16h a, v16h b, v8f c) {
  return __builtin_amdgcn_wmma_f32_16x16x32_f16(
      /*neg_a=*/false, a, /*neg_b=*/false, b,
      /*c_mod=*/(short)0, c, /*reuse_a=*/false, /*reuse_b=*/false);
}

// A fragment 16x32 (MxK), row-major source with leading dim `ld` (caller
// pre-offsets to (row0,k0)). Lane l: row l&15, K runs [kb..kb+7],[kb+16..kb+23],
// kb=(l>>4)*8 -> two 16B contiguous loads.
static __device__ __forceinline__ v16h load_a_frag(const _Float16* base, int ld, int lane) {
  const int r  = lane & 15;
  const int kb = (lane >> 4) << 3;
  const _Float16* p = base + r * ld + kb;
  v8h lo = *(const v8h*)(p);
  v8h hi = *(const v8h*)(p + 16);
  v16h a;
#pragma unroll
  for (int i = 0; i < 8; ++i) { a[i] = lo[i]; a[i + 8] = hi[i]; }
  return a;
}

// B fragment 32x16 (KxN) from an N-major buffer: B[k][n] = baseT[n*ldt + k].
// Lane l: column l&15, K runs [kb..kb+7],[kb+16..kb+23], kb=(l>>4)*8 -> two 16B loads.
static __device__ __forceinline__ v16h load_b_frag(const _Float16* baseT, int ldt, int lane) {
  const int n  = lane & 15;
  const int kb = (lane >> 4) << 3;
  const _Float16* p = baseT + n * ldt + kb;
  v8h lo = *(const v8h*)(p);
  v8h hi = *(const v8h*)(p + 16);
  v16h b;
#pragma unroll
  for (int i = 0; i < 8; ++i) { b[i] = lo[i]; b[i + 8] = hi[i]; }
  return b;
}

// ---------------- conversion kernels ----------------
__global__ void k_cvt_f16(const float* __restrict__ src, _Float16* __restrict__ dst, int n) {
  int i = blockIdx.x * blockDim.x + threadIdx.x;
  if (i < n) dst[i] = (_Float16)src[i];
}

// w: [H][DMODEL][DK] (f32) -> wt: [H][DK][DMODEL] (f16)
__global__ void k_cvt_w_t(const float* __restrict__ w, _Float16* __restrict__ wt) {
  int i = blockIdx.x * blockDim.x + threadIdx.x;      // H*DMODEL*DK total
  int h = i / (DMODEL * DK);
  int r = (i / DK) % DMODEL;
  int c = i % DK;
  wt[(size_t)h * DK * DMODEL + (size_t)c * DMODEL + r] = (_Float16)w[i];
}

// ---------------- QKV projection ----------------
// grid = (BN/16, HEADS), block = 32
__global__ void __launch_bounds__(32)
k_qkv(const _Float16* __restrict__ Xh,
      const _Float16* __restrict__ Wq,
      const _Float16* __restrict__ Wk,
      const _Float16* __restrict__ Wv,
      const float* __restrict__ qb,
      const float* __restrict__ kbias,
      const float* __restrict__ vb,
      _Float16* __restrict__ Qh,
      _Float16* __restrict__ Kh,
      _Float16* __restrict__ Vt) {
  const int mt   = blockIdx.x;
  const int h    = blockIdx.y;
  const int lane = threadIdx.x & 31;
  const int g0   = mt * 16;
  const int b    = g0 >> 11;           // /SEQ (tiles never straddle batches)
  const int n0   = g0 & (SEQ - 1);

  const _Float16* xrow = Xh + (size_t)g0 * DMODEL;
  const _Float16* wqh  = Wq + (size_t)h * DK * DMODEL;
  const _Float16* wkh  = Wk + (size_t)h * DK * DMODEL;
  const _Float16* wvh  = Wv + (size_t)h * DK * DMODEL;

  v8f aq[4], ak[4], av[4];
#pragma unroll
  for (int t = 0; t < 4; ++t) { aq[t] = vzero8(); ak[t] = vzero8(); av[t] = vzero8(); }

#pragma unroll 4
  for (int ks = 0; ks < DMODEL / 32; ++ks) {
    const int k0 = ks * 32;
    if (ks + 1 < DMODEL / 32)
      __builtin_prefetch((const void*)(xrow + k0 + 32), 0, 3);
    v16h a = load_a_frag(xrow + k0, DMODEL, lane);
#pragma unroll
    for (int t = 0; t < 4; ++t) {
      v16h bq = load_b_frag(wqh + (size_t)(t * 16) * DMODEL + k0, DMODEL, lane);
      aq[t] = wmma_f16(a, bq, aq[t]);
      v16h bk = load_b_frag(wkh + (size_t)(t * 16) * DMODEL + k0, DMODEL, lane);
      ak[t] = wmma_f16(a, bk, ak[t]);
      v16h bv = load_b_frag(wvh + (size_t)(t * 16) * DMODEL + k0, DMODEL, lane);
      av[t] = wmma_f16(a, bv, av[t]);
    }
  }

  const int mrow  = (lane >> 4) << 3;
  const int col16 = lane & 15;
  const int bh    = b * HEADS + h;
  _Float16* Qrow = Qh + ((size_t)bh * SEQ + n0) * DK;
  _Float16* Krow = Kh + ((size_t)bh * SEQ + n0) * DK;
  _Float16* Vth  = Vt + (size_t)bh * DK * SEQ;   // [DK][SEQ] per (b,h)

#pragma unroll
  for (int t = 0; t < 4; ++t) {
#pragma unroll
    for (int j = 0; j < 8; ++j) {
      const int m  = j + mrow;
      const int dk = t * 16 + col16;
      Qrow[(size_t)m * DK + dk] = (_Float16)(aq[t][j] + qb[h * DK + dk]);
      Krow[(size_t)m * DK + dk] = (_Float16)(ak[t][j] + kbias[h * DK + dk]);
      Vth[(size_t)dk * SEQ + n0 + m] = (_Float16)(av[t][j] + vb[h * DK + dk]);
    }
  }
}

// ---------------- flash attention (async LDS double-buffered K/V) ----------------
// grid = (BATCH*HEADS, SEQ/64), block = 128 (4 waves, one 16-query tile each)

// per-thread slice of the K (32x64 f16, 4KB) and V^T (64x32 f16, 4KB) tiles:
// 32B from each via 2x global_load_async_to_lds_b128 (ASYNCcnt-tracked DMA).
static __device__ __forceinline__ void stage_kv_async(const _Float16* Kg,   // += key0*DK
                                                      const _Float16* Vg,   // += key0
                                                      _Float16* kbuf, _Float16* vbuf,
                                                      int tid) {
  unsigned long long kga = (unsigned long long)(uintptr_t)Kg + (unsigned)tid * 32u;
  unsigned           kla = (unsigned)(uintptr_t)kbuf + (unsigned)tid * 32u;
  asm volatile("global_load_async_to_lds_b128 %0, %1, off\n\t"
               "global_load_async_to_lds_b128 %0, %1, off offset:16"
               :: "v"(kla), "v"(kga) : "memory");
  const int r = tid >> 1, hcol = (tid & 1) * 32;   // row of V^T, 32B half
  unsigned long long vga = (unsigned long long)(uintptr_t)Vg
                         + (unsigned long long)r * (SEQ * 2) + (unsigned)hcol;
  unsigned           vla = (unsigned)(uintptr_t)vbuf + (unsigned)(r * 64 + hcol);
  asm volatile("global_load_async_to_lds_b128 %0, %1, off\n\t"
               "global_load_async_to_lds_b128 %0, %1, off offset:16"
               :: "v"(vla), "v"(vga) : "memory");
}

static __device__ __forceinline__ void wait_async0() {
  asm volatile("s_wait_asynccnt 0x0" ::: "memory");
}

__global__ void __launch_bounds__(128)
k_attn(const _Float16* __restrict__ Qh,
       const _Float16* __restrict__ Kh,
       const _Float16* __restrict__ Vt,
       _Float16* __restrict__ Ah) {
  const int bh   = blockIdx.x;
  const int tid  = threadIdx.x;
  const int wave = tid >> 5;
  const int lane = tid & 31;
  const int b    = bh >> 3;
  const int h    = bh & 7;
  const int q0   = blockIdx.y * 64 + wave * 16;

  const _Float16* Qbase = Qh + ((size_t)bh * SEQ + q0) * DK;
  const _Float16* Kbase = Kh + (size_t)bh * SEQ * DK;
  const _Float16* Vbase = Vt + (size_t)bh * DK * SEQ;

  __shared__ __attribute__((aligned(128))) _Float16 Kbuf[2][32 * DK];   // [key][dk]
  __shared__ __attribute__((aligned(128))) _Float16 Vbuf[2][DK * 32];   // [dk][key]
  __shared__ __attribute__((aligned(128))) _Float16 Pbuf[4][16 * 32];   // per-wave

  // Q tile A-fragments, resident for the whole sweep
  const v16h aQ0 = load_a_frag(Qbase + 0,  DK, lane);
  const v16h aQ1 = load_a_frag(Qbase + 32, DK, lane);

  float m_run[8], l_run[8];
  v8f accO[4];
#pragma unroll
  for (int j = 0; j < 8; ++j) { m_run[j] = -3.0e38f; l_run[j] = 0.f; }
#pragma unroll
  for (int t = 0; t < 4; ++t) accO[t] = vzero8();

  const float scale = 0.125f;          // 1/sqrt(DK)
  const int   mrow  = (lane >> 4) << 3;
  const int   col16 = lane & 15;
  _Float16* Pw = &Pbuf[wave][0];

  // prologue: stage tile 0
  stage_kv_async(Kbase, Vbase, Kbuf[0], Vbuf[0], tid);

  for (int kt = 0; kt < SEQ / 32; ++kt) {
    const int cur = kt & 1;
    wait_async0();        // my 4 async ops for tile kt landed
    __syncthreads();      // everyone's landed; everyone done reading buf[1-cur]

    if (kt + 1 < SEQ / 32) {  // overlap next tile's DMA with this tile's math
      const int key1 = (kt + 1) * 32;
      stage_kv_async(Kbase + (size_t)key1 * DK, Vbase + key1,
                     Kbuf[1 - cur], Vbuf[1 - cur], tid);
    }

    const _Float16* Kb = Kbuf[cur];
    const _Float16* Vb = Vbuf[cur];

    // S = Q K^T for 32 keys (two 16-wide column tiles), all from LDS
    v8f s0 = vzero8(), s1 = vzero8();
    s0 = wmma_f16(aQ0, load_b_frag(Kb,                DK, lane), s0);
    s0 = wmma_f16(aQ1, load_b_frag(Kb + 32,           DK, lane), s0);
    s1 = wmma_f16(aQ0, load_b_frag(Kb + 16 * DK,      DK, lane), s1);
    s1 = wmma_f16(aQ1, load_b_frag(Kb + 16 * DK + 32, DK, lane), s1);

    // online softmax; row j+mrow lives across a 16-lane half in VGPR j
    float p0[8], p1[8], alpha[8];
#pragma unroll
    for (int j = 0; j < 8; ++j) {
      float v0 = s0[j] * scale;
      float v1 = s1[j] * scale;
      float mx = fmaxf(v0, v1);
      mx = fmaxf(mx, __shfl_xor(mx, 1, 32));
      mx = fmaxf(mx, __shfl_xor(mx, 2, 32));
      mx = fmaxf(mx, __shfl_xor(mx, 4, 32));
      mx = fmaxf(mx, __shfl_xor(mx, 8, 32));
      const float rm = fmaxf(m_run[j], mx);
      const float e0 = __expf(v0 - rm);
      const float e1 = __expf(v1 - rm);
      float ls = e0 + e1;
      ls += __shfl_xor(ls, 1, 32);
      ls += __shfl_xor(ls, 2, 32);
      ls += __shfl_xor(ls, 4, 32);
      ls += __shfl_xor(ls, 8, 32);
      alpha[j] = __expf(m_run[j] - rm);
      l_run[j] = l_run[j] * alpha[j] + ls;
      m_run[j] = rm;
      p0[j] = e0;
      p1[j] = e1;
    }
#pragma unroll
    for (int t = 0; t < 4; ++t)
#pragma unroll
      for (int j = 0; j < 8; ++j) accO[t][j] *= alpha[j];

    // re-layout P (C-frag f32) -> A-frag f16 via per-wave LDS scratch
    // (same-wave DS ops are in-order; compiler inserts the dscnt wait)
#pragma unroll
    for (int j = 0; j < 8; ++j) {
      const int m = j + mrow;
      Pw[m * 32 + col16]      = (_Float16)p0[j];
      Pw[m * 32 + 16 + col16] = (_Float16)p1[j];
    }
    const v16h aP = load_a_frag(Pw, 32, lane);

    // O += P * V   (V^T tile: B[k=key][n=dk] = Vb[dk*32 + key])
#pragma unroll
    for (int t = 0; t < 4; ++t) {
      v16h bv = load_b_frag(Vb + (t * 16) * 32, 32, lane);
      accO[t] = wmma_f16(aP, bv, accO[t]);
    }
  }

  // epilogue: divide by row sum, write permuted feature order f = dk*HEADS + h
  float inv[8];
#pragma unroll
  for (int j = 0; j < 8; ++j) inv[j] = 1.0f / l_run[j];
#pragma unroll
  for (int t = 0; t < 4; ++t) {
#pragma unroll
    for (int j = 0; j < 8; ++j) {
      const int m  = j + mrow;
      const int dk = t * 16 + col16;
      const size_t row = (size_t)(b * SEQ + q0 + m);
      Ah[row * DMODEL + dk * HEADS + h] = (_Float16)(accO[t][j] * inv[j]);
    }
  }
}

// ---------------- output projection ----------------
// Y[g][o] = sum_f Ah[g][f] * out_w[o][f]; out_w rows are already BT layout.
// grid = (BN/16, DMODEL/64), block = 32
__global__ void __launch_bounds__(32)
k_outproj(const _Float16* __restrict__ Ah,
          const _Float16* __restrict__ OWh,
          float* __restrict__ Y) {
  const int mt   = blockIdx.x;
  const int nc   = blockIdx.y;
  const int lane = threadIdx.x & 31;
  const _Float16* arow = Ah + (size_t)mt * 16 * DMODEL;

  v8f acc[4];
#pragma unroll
  for (int t = 0; t < 4; ++t) acc[t] = vzero8();

#pragma unroll 4
  for (int ks = 0; ks < DMODEL / 32; ++ks) {
    const int k0 = ks * 32;
    v16h a = load_a_frag(arow + k0, DMODEL, lane);
#pragma unroll
    for (int t = 0; t < 4; ++t) {
      v16h bw = load_b_frag(OWh + (size_t)(nc * 64 + t * 16) * DMODEL + k0, DMODEL, lane);
      acc[t] = wmma_f16(a, bw, acc[t]);
    }
  }

  const int mrow  = (lane >> 4) << 3;
  const int col16 = lane & 15;
#pragma unroll
  for (int t = 0; t < 4; ++t) {
#pragma unroll
    for (int j = 0; j < 8; ++j) {
      const int m   = j + mrow;
      const int col = nc * 64 + t * 16 + col16;
      Y[((size_t)mt * 16 + m) * DMODEL + col] = acc[t][j];
    }
  }
}

// ---------------- launcher ----------------
extern "C" void kernel_launch(void* const* d_in, const int* in_sizes, int n_in,
                              void* d_out, int out_size, void* d_ws, size_t ws_size,
                              hipStream_t stream) {
  const float* latent = (const float*)d_in[0];  // [4,2048,512]
  const float* q_proj = (const float*)d_in[1];  // [8,512,64]
  const float* k_proj = (const float*)d_in[2];
  const float* v_proj = (const float*)d_in[3];
  const float* q_bias = (const float*)d_in[4];  // [8,64]
  const float* k_bias = (const float*)d_in[5];
  const float* v_bias = (const float*)d_in[6];
  const float* out_w  = (const float*)d_in[7];  // [512,512]
  float* out = (float*)d_out;

  // fp16 workspace carving (~44 MB)
  const size_t XH = (size_t)BN * DMODEL;          // 4194304 (also Q/K/V/A size)
  const size_t WH = (size_t)HEADS * DK * DMODEL;  // 262144
  const size_t OW = (size_t)DMODEL * DMODEL;      // 262144
  _Float16* ws  = (_Float16*)d_ws;
  _Float16* Xh  = ws;
  _Float16* Wq  = Xh  + XH;
  _Float16* Wk  = Wq  + WH;
  _Float16* Wv  = Wk  + WH;
  _Float16* OWh = Wv  + WH;
  _Float16* Qh  = OWh + OW;
  _Float16* Kh  = Qh  + XH;
  _Float16* Vt  = Kh  + XH;
  _Float16* Ah  = Vt  + XH;
  (void)in_sizes; (void)n_in; (void)out_size; (void)ws_size;

  // 1) fp32 -> fp16 conversions / weight transposes
  k_cvt_f16<<<(int)(XH / 256), 256, 0, stream>>>(latent, Xh, (int)XH);
  k_cvt_f16<<<(int)(OW / 256), 256, 0, stream>>>(out_w, OWh, (int)OW);
  k_cvt_w_t<<<(int)(WH / 256), 256, 0, stream>>>(q_proj, Wq);
  k_cvt_w_t<<<(int)(WH / 256), 256, 0, stream>>>(k_proj, Wk);
  k_cvt_w_t<<<(int)(WH / 256), 256, 0, stream>>>(v_proj, Wv);

  // 2) QKV projection (WMMA)
  k_qkv<<<dim3(BN / 16, HEADS), 32, 0, stream>>>(Xh, Wq, Wk, Wv,
                                                 q_bias, k_bias, v_bias,
                                                 Qh, Kh, Vt);

  // 3) flash attention (WMMA + async LDS DMA + online softmax)
  k_attn<<<dim3(BATCH * HEADS, SEQ / 64), 128, 0, stream>>>(Qh, Kh, Vt, Ah);

  // 4) output projection (WMMA), f32 result
  k_outproj<<<dim3(BN / 16, DMODEL / 64), 32, 0, stream>>>(Ah, OWh, out);
}
